// MultiHeadAttention_7318624272718
// MI455X (gfx1250) — matrix operand order repaired and checked by reading, not compiled, not run on hardware
//
#include <hip/hip_runtime.h>

#define DEV __device__ __forceinline__

typedef __bf16 bf16_t;
typedef __attribute__((ext_vector_type(16))) __bf16 v16bf;
typedef __attribute__((ext_vector_type(8)))  __bf16 v8bf;
typedef __attribute__((ext_vector_type(8)))  float   v8f;
typedef unsigned short u16;
typedef unsigned int   u32;

constexpr int EMB   = 1024;
constexpr int HEADS = 4;
constexpr int HD    = 256;   // head dim
constexpr int BATCH = 4;
constexpr int SEQ   = 2048;
constexpr int MROWS = BATCH * SEQ; // 8192

// ---------- helpers ----------

DEV u16 f32_to_bf16(float f) {
  union { float f; u32 u; } v; v.f = f;
  u32 u = v.u;
  u32 r = u + 0x7FFFu + ((u >> 16) & 1u);   // round-to-nearest-even
  return (u16)(r >> 16);
}

// A-matrix (16x32 bf16) per-lane fragment: elements 0..15 = K(ko..ko+15)
// (ko = (lane>>4)*16), matching the contiguous B fragments.
DEV v16bf make_a(const bf16_t* p0, const bf16_t* p1) {
  v8bf lo = *(const v8bf*)p0;
  v8bf hi = *(const v8bf*)p1;
  return __builtin_shufflevector(lo, hi, 0,1,2,3,4,5,6,7,8,9,10,11,12,13,14,15);
}

DEV v8f wmma_bf16(v16bf a, v16bf b, v8f c) {
  return __builtin_amdgcn_wmma_f32_16x16x32_bf16(false, a, false, b, (short)0, c,
                                                 false, false);
}

// gfx1250 async copy global -> LDS (16B per lane), tracked by ASYNCcnt.
// lds_byte = wave-relative LDS byte offset (low 32 bits of generic pointer).
DEV void async_b128(u32 lds_byte, const void* g) {
  asm volatile("global_load_async_to_lds_b128 %0, %1, off"
               :: "v"(lds_byte), "v"(g)
               : "memory");
}
DEV void wait_async0() {
  asm volatile("s_wait_asynccnt 0x0" ::: "memory");
}
DEV u32 lds_off(const void* p) { return (u32)(size_t)p; }

// ---------- prologue conversions ----------

__global__ void cvt_f32_bf16(const float* __restrict__ in, u16* __restrict__ out, int n) {
  int i = blockIdx.x * blockDim.x + threadIdx.x;
  if (i < n) out[i] = f32_to_bf16(in[i]);
}

// in: [rows, cols] f32 row-major -> out: [cols, rows] bf16 (W -> Wt[N][K])
__global__ void transpose_to_bf16(const float* __restrict__ in, u16* __restrict__ out,
                                  int rows, int cols) {
  int i = blockIdx.x * blockDim.x + threadIdx.x;
  if (i < rows * cols) {
    int r = i / cols, c = i % cols;
    out[(size_t)c * rows + r] = f32_to_bf16(in[i]);
  }
}

// ---------- WMMA GEMM: C[M,N] = A[M,K](bf16) * Bt[N,K]^T(bf16) + bias ----------
// Block = 128x128 tile, 8 waves of 16x128 each. B tile (128 cols x 32 K, 8KB)
// is double-buffered in LDS via async global->LDS copies shared by all waves.

template<bool OUT_F32, bool V_TRANS>
__global__ void __launch_bounds__(256)
gemm_bias_wmma(const u16* __restrict__ Abf,   // [M,K] bf16
               const u16* __restrict__ Btbf,  // [N,K] bf16 (B transposed)
               const float* __restrict__ bias,
               u16* __restrict__ Cb, float* __restrict__ Cf,
               int M, int N, int K) {
  __shared__ __align__(32) u16 ldsB[2][128 * 32];   // [buf][col][k] 8KB each
  int tid  = threadIdx.x;
  int wave = tid >> 5, lane = tid & 31;
  int nb = N >> 7;
  int bm = blockIdx.x / nb, bn = blockIdx.x % nb;
  int m0 = bm * 128 + wave * 16, n0 = bn * 128;
  int row = lane & 15, half = lane >> 4;

  const bf16_t* Arow = (const bf16_t*)Abf + (size_t)(m0 + row) * K;

  // async staging map: thread -> 32B of the B tile (col = tid/2, half-row = tid&1)
  int bcol = tid >> 1, bpart = tid & 1;
  const char* Bg = (const char*)(Btbf + (size_t)(bn * 128 + bcol) * K + bpart * 16);
  u32 lb[2] = { lds_off(&ldsB[0][bcol * 32 + bpart * 16]),
                lds_off(&ldsB[1][bcol * 32 + bpart * 16]) };
  auto kickB = [&](int buf, int kc) {
    const char* g = Bg + (size_t)kc * 64;   // 32 halves along K = 64B
    async_b128(lb[buf],      g);
    async_b128(lb[buf] + 16, g + 16);
  };

  v8f zero = {0.f,0.f,0.f,0.f,0.f,0.f,0.f,0.f};
  v8f acc[8];
#pragma unroll
  for (int t = 0; t < 8; ++t) acc[t] = zero;

  const int KC = K >> 5;
  kickB(0, 0);
  for (int kc = 0; kc < KC; ++kc) {
    int buf = kc & 1;
    wait_async0();
    __syncthreads();
    if (kc + 1 < KC) kickB(buf ^ 1, kc + 1);

    const bf16_t* Bl = (const bf16_t*)ldsB[buf];
    v16bf a = make_a(Arow + kc * 32 + half * 16, Arow + kc * 32 + 8 + half * 16);
    // batch all fragment loads first (one DS clause), then the WMMA chain
    v16bf bfrag[8];
#pragma unroll
    for (int t = 0; t < 8; ++t)
      bfrag[t] = *(const v16bf*)(Bl + (t * 16 + row) * 32 + half * 16);
#pragma unroll
    for (int t = 0; t < 8; ++t)
      acc[t] = wmma_bf16(a, bfrag[t], acc[t]);
    __syncthreads();
  }

#pragma unroll
  for (int t = 0; t < 8; ++t) {
    int n = n0 + t * 16 + row;
    float bv = bias[n];
#pragma unroll
    for (int g = 0; g < 8; ++g) {
      int m = m0 + half * 8 + g;            // C layout: lanes>=16 hold rows 8..15
      float val = acc[t][g] + bv;
      if constexpr (OUT_F32) {
        Cf[(size_t)m * N + n] = val;
      } else if constexpr (V_TRANS) {       // write V as Vt[B,H,D,S]
        int b_ = m >> 11, s = m & 2047;     // SEQ = 2048
        int hh = n >> 8,  d = n & 255;      // HD  = 256
        Cb[((((size_t)b_ * HEADS + hh) * HD + d) << 11) + s] = f32_to_bf16(val);
      } else {
        Cb[(size_t)m * N + n] = f32_to_bf16(val);
      }
    }
  }
}

// ---------- flash attention ----------
// Block = 8 waves = 8 consecutive 16-row q-tiles of the SAME (b,h); K and V
// kv-chunks (32 x 256 bf16 = 16KB each) are double-buffered in LDS via async
// copies and shared by all 8 waves.

__global__ void __launch_bounds__(256)
attn_flash(const u16* __restrict__ Qb,   // [B,S,E] bf16
           const u16* __restrict__ Kb,   // [B,S,E] bf16
           const u16* __restrict__ Vt,   // [B,H,D,S] bf16
           u16* __restrict__ Ab) {       // [B,S,E] bf16 out
  __shared__ __align__(32) u16 ldsK[2][32 * 256];   // [kv][dim]   16KB each
  __shared__ __align__(32) u16 ldsV[2][256 * 32];   // [dim][kv]   16KB each
  __shared__ __align__(32) u16 ldsP[8][16 * 32];    // P bounce, 1KB/wave

  int tid  = threadIdx.x;
  int warp = tid >> 5, lane = tid & 31;
  int qt  = blockIdx.x * 8 + warp;        // 2048 q-tiles total
  int b   = qt >> 9;                      // / (HEADS * SEQ/16)
  int rem = qt & 511;
  int h   = rem >> 7;                     // same for all 8 waves in block
  int q0  = (rem & 127) << 4;
  int row = lane & 15, half = lane >> 4;

  const bf16_t* Q   = (const bf16_t*)Qb + (size_t)(b * SEQ + q0 + row) * EMB + h * HD;
  const char*   Kg  = (const char*)(Kb + (size_t)(b * SEQ) * EMB + h * HD);
  const char*   Vg  = (const char*)(Vt + (size_t)((b * HEADS + h) * HD) * SEQ);
  u16* pl = ldsP[warp];

  u32 lK[2] = { lds_off(&ldsK[0][0]), lds_off(&ldsK[1][0]) };
  u32 lV[2] = { lds_off(&ldsV[0][0]), lds_off(&ldsV[1][0]) };

  // each of 256 threads stages 64B of K and 64B of V per chunk (4x b128 each)
  auto kickKV = [&](int buf, int kv0n) {
    const char* gk = Kg + (size_t)kv0n * (EMB * 2);
    const char* gv = Vg + (size_t)kv0n * 2;
#pragma unroll
    for (int j = 0; j < 4; ++j) {
      int byt = tid * 64 + j * 16;              // 0..16383
      int kvr = byt >> 9, offk = byt & 511;     // K rows are 512B
      async_b128(lK[buf] + byt, gk + (size_t)kvr * (EMB * 2) + offk);
      int dim = byt >> 6, offv = byt & 63;      // V rows are 64B
      async_b128(lV[buf] + byt, gv + (size_t)dim * (SEQ * 2) + offv);
    }
  };

  v16bf qa[8];                            // Q tile, 8 K-chunks of 32
#pragma unroll
  for (int c = 0; c < 8; ++c)
    qa[c] = make_a(Q + c * 32 + half * 16, Q + c * 32 + 8 + half * 16);

  v8f zero = {0.f,0.f,0.f,0.f,0.f,0.f,0.f,0.f};
  v8f oacc[16];                           // 16x256 f32 accumulator
#pragma unroll
  for (int t = 0; t < 16; ++t) oacc[t] = zero;
  float mrow[8], lsum[8];
#pragma unroll
  for (int g = 0; g < 8; ++g) { mrow[g] = -1e30f; lsum[g] = 0.f; }

  const float kSc = 1.4426950408889634f * 0.0625f;   // log2(e)/sqrt(HD)

  kickKV(0, 0);
  for (int kv0 = 0; kv0 < SEQ; kv0 += 32) {
    int buf = (kv0 >> 5) & 1;
    wait_async0();
    __syncthreads();
    if (kv0 + 32 < SEQ) kickKV(buf ^ 1, kv0 + 32);

    const bf16_t* Kl = (const bf16_t*)ldsK[buf];
    const bf16_t* Vl = (const bf16_t*)ldsV[buf];

    // batch all K fragments (one DS clause), then the 16-deep WMMA chain
    v16bf kb0[8], kb1[8];
#pragma unroll
    for (int c = 0; c < 8; ++c) {
      kb0[c] = *(const v16bf*)(Kl + row * 256 + c * 32 + half * 16);
      kb1[c] = *(const v16bf*)(Kl + (row + 16) * 256 + c * 32 + half * 16);
    }
    v8f s0 = zero, s1 = zero;
#pragma unroll
    for (int c = 0; c < 8; ++c) {
      s0 = wmma_bf16(qa[c], kb0[c], s0);
      s1 = wmma_bf16(qa[c], kb1[c], s1);
    }

    // issue V fragment loads early so DS latency hides under the softmax math
    v16bf vb[16];
#pragma unroll
    for (int t = 0; t < 16; ++t)
      vb[t] = *(const v16bf*)(Vl + (t * 16 + row) * 32 + half * 16);

    // online softmax; row stats live in the lanes matching the C layout
    float corr[8];
#pragma unroll
    for (int g = 0; g < 8; ++g) {
      float a0 = s0[g] * kSc, a1 = s1[g] * kSc;
      float mx = fmaxf(a0, a1);
#pragma unroll
      for (int msk = 8; msk >= 1; msk >>= 1)
        mx = fmaxf(mx, __shfl_xor(mx, msk));          // reduce within 16-lane half
      float mnew = fmaxf(mrow[g], mx);
      float p0 = exp2f(a0 - mnew);
      float p1 = exp2f(a1 - mnew);
      float cr = exp2f(mrow[g] - mnew);
      mrow[g] = mnew;
      lsum[g] = lsum[g] * cr + p0 + p1;
      corr[g] = cr;
      int r = half * 8 + g;               // C-layout row for this lane
      pl[r * 32 + row]      = f32_to_bf16(p0);
      pl[r * 32 + 16 + row] = f32_to_bf16(p1);
    }

    // re-read P in A-matrix layout (wave-synchronous LDS, DScnt-ordered)
    v16bf pa = make_a((const bf16_t*)pl + row * 32 + half * 16,
                      (const bf16_t*)pl + row * 32 + 8 + half * 16);

    // O = diag(corr)*O + P * V
#pragma unroll
    for (int t = 0; t < 16; ++t) {
      v8f o = oacc[t];
#pragma unroll
      for (int g = 0; g < 8; ++g) o[g] *= corr[g];
      oacc[t] = wmma_bf16(pa, vb[t], o);
    }
    __syncthreads();
  }

  float linv[8];
#pragma unroll
  for (int g = 0; g < 8; ++g) {
    float l = lsum[g];
#pragma unroll
    for (int msk = 8; msk >= 1; msk >>= 1)
      l += __shfl_xor(l, msk);
    linv[g] = 1.0f / l;
  }
#pragma unroll
  for (int t = 0; t < 16; ++t) {
#pragma unroll
    for (int g = 0; g < 8; ++g) {
      int m = q0 + half * 8 + g;
      int n = h * HD + t * 16 + row;
      Ab[(size_t)(b * SEQ + m) * EMB + n] = f32_to_bf16(oacc[t][g] * linv[g]);
    }
  }
}

// ---------- host ----------

extern "C" void kernel_launch(void* const* d_in, const int* in_sizes, int n_in,
                              void* d_out, int out_size, void* d_ws, size_t ws_size,
                              hipStream_t stream) {
  (void)in_sizes; (void)n_in; (void)out_size; (void)ws_size;
  const float* x  = (const float*)d_in[0];
  const float* Wq = (const float*)d_in[1];
  const float* bq = (const float*)d_in[2];
  const float* Wk = (const float*)d_in[3];
  const float* bk = (const float*)d_in[4];
  const float* Wv = (const float*)d_in[5];
  const float* bv = (const float*)d_in[6];
  const float* Wo = (const float*)d_in[7];
  const float* bo = (const float*)d_in[8];
  float* out = (float*)d_out;

  char* ws = (char*)d_ws;
  size_t off = 0;
  auto alloc = [&](size_t bytes) { char* p = ws + off; off += bytes; return p; };
  u16* xb  = (u16*)alloc((size_t)MROWS * EMB * 2);   // 16 MiB
  u16* WqT = (u16*)alloc((size_t)EMB * EMB * 2);     //  2 MiB
  u16* WkT = (u16*)alloc((size_t)EMB * EMB * 2);
  u16* WvT = (u16*)alloc((size_t)EMB * EMB * 2);
  u16* WoT = (u16*)alloc((size_t)EMB * EMB * 2);
  u16* Qbf = (u16*)alloc((size_t)MROWS * EMB * 2);   // 16 MiB
  u16* Kbf = (u16*)alloc((size_t)MROWS * EMB * 2);
  u16* Vtb = (u16*)alloc((size_t)MROWS * EMB * 2);   // [B,H,D,S]
  u16* Abf = (u16*)alloc((size_t)MROWS * EMB * 2);   // attn out, total 88 MiB

  int nx = MROWS * EMB;
  cvt_f32_bf16<<<nx / 256, 256, 0, stream>>>(x, xb, nx);
  int nw = EMB * EMB;
  transpose_to_bf16<<<nw / 256, 256, 0, stream>>>(Wq, WqT, EMB, EMB);
  transpose_to_bf16<<<nw / 256, 256, 0, stream>>>(Wk, WkT, EMB, EMB);
  transpose_to_bf16<<<nw / 256, 256, 0, stream>>>(Wv, WvT, EMB, EMB);
  transpose_to_bf16<<<nw / 256, 256, 0, stream>>>(Wo, WoT, EMB, EMB);

  // (M/128)*(N/128) = 64*8 = 512 blocks of 8 waves
  int gemm_blocks = (MROWS / 128) * (EMB / 128);
  gemm_bias_wmma<false, false><<<gemm_blocks, 256, 0, stream>>>(
      xb, WqT, bq, Qbf, nullptr, MROWS, EMB, EMB);
  gemm_bias_wmma<false, false><<<gemm_blocks, 256, 0, stream>>>(
      xb, WkT, bk, Kbf, nullptr, MROWS, EMB, EMB);
  gemm_bias_wmma<false, true><<<gemm_blocks, 256, 0, stream>>>(
      xb, WvT, bv, Vtb, nullptr, MROWS, EMB, EMB);

  // 2048 q-tiles / 8 waves per block = 256 blocks
  attn_flash<<<256, 256, 0, stream>>>(Qbf, Kbf, Vtb, Abf);

  gemm_bias_wmma<true, false><<<gemm_blocks, 256, 0, stream>>>(
      Abf, WoT, bo, nullptr, out, MROWS, EMB, EMB);
}
